// HCAAttention_88716844466583
// MI455X (gfx1250) — compile-verified, matrix operand order checked by reading, not run on hardware
//
#include <hip/hip_runtime.h>
#include <hip/hip_bf16.h>
#include <math.h>

// ---------------------------------------------------------------------------
// HCA attention for MI455X (gfx1250, wave32, WMMA bf16 16x16x32, f32 accum)
// B=2 L=2048 D=1024 H=16 hd=64
// ---------------------------------------------------------------------------

typedef __attribute__((ext_vector_type(16))) __bf16 v16bf;
typedef __attribute__((ext_vector_type(8)))  __bf16 v8bf;
typedef __attribute__((ext_vector_type(8)))  float  v8f;

#define LQ   2048
#define DD   1024
#define HH   16
#define HDIM 64
#define SROW 2052           // padded LDS score-row stride (floats), 16B aligned

__device__ __forceinline__ v8f wmma_bf16(v16bf a, v16bf b, v8f c) {
  return __builtin_amdgcn_wmma_f32_16x16x32_bf16(false, a, false, b,
                                                 (short)0, c, false, false);
}

// Load a 16-element bf16 operand run-pair (runs of 8 contiguous elems).
__device__ __forceinline__ v16bf ld16_bf(const __bf16* p, const __bf16* q) {
  v8bf a = *(const v8bf*)p;
  v8bf b = *(const v8bf*)q;
  v16bf v;
#pragma unroll
  for (int i = 0; i < 8; ++i) { v[i] = a[i]; v[8 + i] = b[i]; }
  return v;
}

// Load 16 fp32 (two runs of 8) and convert to bf16 operand.
__device__ __forceinline__ v16bf ld16_f32(const float* p, const float* q) {
  float4 a0 = *(const float4*)p;
  float4 a1 = *(const float4*)(p + 4);
  float4 b0 = *(const float4*)q;
  float4 b1 = *(const float4*)(q + 4);
  v16bf v;
  v[0]  = (__bf16)a0.x; v[1]  = (__bf16)a0.y; v[2]  = (__bf16)a0.z; v[3]  = (__bf16)a0.w;
  v[4]  = (__bf16)a1.x; v[5]  = (__bf16)a1.y; v[6]  = (__bf16)a1.z; v[7]  = (__bf16)a1.w;
  v[8]  = (__bf16)b0.x; v[9]  = (__bf16)b0.y; v[10] = (__bf16)b0.z; v[11] = (__bf16)b0.w;
  v[12] = (__bf16)b1.x; v[13] = (__bf16)b1.y; v[14] = (__bf16)b1.z; v[15] = (__bf16)b1.w;
  return v;
}

// ---------------------------------------------------------------------------
// Kernel 1: QKV projection.  y = x @ W^T + b.  grid.z selects Q/K/V.
// Each wave computes one 16x16 output tile, K-loop in steps of 32.
// Q,K stored bf16 row-major (token, D) in ws; V stored bf16 transposed
// (b,h,d,token) in ws + fp32 (b,h,token,d) into d_out's v region.
// ---------------------------------------------------------------------------
__global__ void hca_qkv_kernel(const float* __restrict__ x,
                               const float* __restrict__ fb,
                               const float* __restrict__ Wq, const float* __restrict__ bq,
                               const float* __restrict__ Wk, const float* __restrict__ bk,
                               const float* __restrict__ Wv, const float* __restrict__ bv,
                               __bf16* __restrict__ Qws, __bf16* __restrict__ Kws,
                               __bf16* __restrict__ Vtws, float* __restrict__ Vout) {
  const int lane  = threadIdx.x & 31;
  const int wave  = threadIdx.x >> 5;
  const int which = blockIdx.z;
  const float* W    = (which == 0) ? Wq : ((which == 1) ? Wk : Wv);
  const float* bias = (which == 0) ? bq : ((which == 1) ? bk : bv);

  const int m0   = blockIdx.y * 16;                 // token tile (flattened B*L)
  const int n0   = (blockIdx.x * 8 + wave) * 16;    // output-dim tile
  const int half = lane >> 4;
  const int lx   = lane & 15;
  const int kbase = half * 8;

  const float* arow = x + (size_t)(m0 + lx) * DD;   // A: row = token, contiguous k
  const float* brow = W + (size_t)(n0 + lx) * DD;   // B: row = out-dim, contiguous k

  v8f acc = {};
  for (int kt = 0; kt < DD; kt += 32) {
    __builtin_prefetch(arow + kt + 128, 0, 0);
    v16bf A = ld16_f32(arow + kt + kbase, arow + kt + 16 + kbase);
    v16bf B = ld16_f32(brow + kt + kbase, brow + kt + 16 + kbase);
    acc = wmma_bf16(A, B, acc);
  }

  const int ncol = n0 + lx;
  float add = bias[ncol];
  const int bidx = m0 >> 11;                        // batch of this token tile
  if (which == 0) add += fb[bidx * DD + ncol];      // cyclic feedback on Q

#pragma unroll
  for (int r = 0; r < 8; ++r) {
    const int rr  = r + half * 8;
    const int tok = m0 + rr;
    const float v = acc[r] + add;
    if (which == 0) {
      Qws[(size_t)tok * DD + ncol] = (__bf16)v;
    } else if (which == 1) {
      Kws[(size_t)tok * DD + ncol] = (__bf16)v;
    } else {
      const int bb = tok >> 11, ti = tok & (LQ - 1);
      const int h = ncol >> 6, d = ncol & (HDIM - 1);
      Vtws[(((size_t)(bb * HH + h) * HDIM + d) << 11) + ti] = (__bf16)v;
      Vout[(((size_t)(bb * HH + h) << 11) + ti) * HDIM + d] = v;
    }
  }
}

// ---------------------------------------------------------------------------
// Kernel 2: attention.  One wave per 16-query tile; 2 waves / block.
// Scores strip (16 x 2048 fp32) lives in LDS; 3 passes:
//   (1) S = QK^T/8 + tree-bias + causal, track row max
//   (2) p = exp(s-m), row sums, acc += P @ V (bf16 WMMA)
//   (3) write normalized fp32 probabilities to d_out (attn region)
// ---------------------------------------------------------------------------
__global__ void hca_attn_kernel(const __bf16* __restrict__ Qws,
                                const __bf16* __restrict__ Kws,
                                const __bf16* __restrict__ Vtws,
                                __bf16* __restrict__ OutH,
                                float* __restrict__ attn_out) {
  extern __shared__ float smem[];
  const int lane = threadIdx.x & 31;
  const int wave = threadIdx.x >> 5;
  const int t  = blockIdx.x * 2 + wave;
  const int bh = t >> 7;                 // 128 query tiles per (b,h)
  const int qt = t & 127;
  const int b = bh >> 4;
  const int h = bh & 15;
  const int q0 = qt * 16;
  const int half = lane >> 4, lx = lane & 15;
  const int kbase = half * 8;

  float* S       = smem + (size_t)wave * (16 * SROW);
  float* statsM  = smem + 2 * (16 * SROW) + wave * 32;
  float* statsSm = statsM + 16;

  // Q operands (16x32 bf16, two k-steps covering hd=64)
  const __bf16* qrow = Qws + (size_t)((b << 11) + q0 + lx) * DD + h * HDIM;
  const v16bf A0 = ld16_bf(qrow + kbase,      qrow + 16 + kbase);
  const v16bf A1 = ld16_bf(qrow + 32 + kbase, qrow + 48 + kbase);

  float rmax[8];
#pragma unroll
  for (int r = 0; r < 8; ++r) rmax[r] = -3.0e38f;

  // ---- pass 1: scores into LDS, running row max --------------------------
  for (int j0 = 0; j0 <= q0; j0 += 16) {
    const __bf16* krow = Kws + (size_t)((b << 11) + j0 + lx) * DD + h * HDIM;
    const v16bf B0 = ld16_bf(krow + kbase,      krow + 16 + kbase);
    const v16bf B1 = ld16_bf(krow + 32 + kbase, krow + 48 + kbase);
    v8f s = {};
    s = wmma_bf16(A0, B0, s);
    s = wmma_bf16(A1, B1, s);
    const int j = j0 + lx;
#pragma unroll
    for (int r = 0; r < 8; ++r) {
      const int rr = r + half * 8;
      const int i  = q0 + rr;
      float sv = s[r] * 0.125f;                      // 1/sqrt(64)
      sv += (j == i || j + 1 == i) ? 0.0f : -5.0f;   // soft-tree bias
      if (j > i) sv = -__builtin_inff();             // causal
      rmax[r] = fmaxf(rmax[r], sv);
      S[rr * SROW + j0 + lx] = sv;
    }
  }
  // pad score columns up to the next multiple of 32 with -inf
  if (q0 + 16 < LQ) {
#pragma unroll
    for (int e = 0; e < 8; ++e)
      S[lx * SROW + q0 + 16 + kbase + e] = -__builtin_inff();
  }
  // row-max reduction across the 16 lanes holding each row
#pragma unroll
  for (int r = 0; r < 8; ++r) {
    float v = rmax[r];
    v = fmaxf(v, __shfl_xor(v, 1, 32));
    v = fmaxf(v, __shfl_xor(v, 2, 32));
    v = fmaxf(v, __shfl_xor(v, 4, 32));
    v = fmaxf(v, __shfl_xor(v, 8, 32));
    rmax[r] = v;
  }
  if (lx == 0) {
#pragma unroll
    for (int r = 0; r < 8; ++r) statsM[half * 8 + r] = rmax[r];
  }
  const float mrow = statsM[lx];          // A-layout: lane owns row lx

  // ---- pass 2: exp, row sums, P @ V --------------------------------------
  v8f acc[4] = {};
  float rsum = 0.0f;
  const int nkeys = q0 + 16;
  const __bf16* vbase = Vtws + ((size_t)bh * HDIM << 11);
  for (int kb = 0; kb < nkeys; kb += 32) {
    float* r1 = &S[lx * SROW + kb + kbase];
    float* r2 = &S[lx * SROW + kb + 16 + kbase];
    float4 p0 = *(float4*)r1;  float4 p1 = *(float4*)(r1 + 4);
    float4 p2 = *(float4*)r2;  float4 p3 = *(float4*)(r2 + 4);
    float pr[16] = {p0.x, p0.y, p0.z, p0.w, p1.x, p1.y, p1.z, p1.w,
                    p2.x, p2.y, p2.z, p2.w, p3.x, p3.y, p3.z, p3.w};
    v16bf Ap;
#pragma unroll
    for (int e = 0; e < 16; ++e) {
      const float p = __expf(pr[e] - mrow);
      rsum += p;
      pr[e] = p;
      Ap[e] = (__bf16)p;
    }
    *(float4*)r1       = make_float4(pr[0],  pr[1],  pr[2],  pr[3]);
    *(float4*)(r1 + 4) = make_float4(pr[4],  pr[5],  pr[6],  pr[7]);
    *(float4*)r2       = make_float4(pr[8],  pr[9],  pr[10], pr[11]);
    *(float4*)(r2 + 4) = make_float4(pr[12], pr[13], pr[14], pr[15]);
#pragma unroll
    for (int nt = 0; nt < 4; ++nt) {
      const __bf16* vrow = vbase + ((size_t)(nt * 16 + lx) << 11) + kb;
      const v16bf Bv = ld16_bf(vrow + kbase, vrow + 16 + kbase);
      acc[nt] = wmma_bf16(Ap, Bv, acc[nt]);
    }
  }
  // row-sum: lanes lx and lx+16 hold the two halves of row lx
  rsum += __shfl_xor(rsum, 16, 32);
  if (half == 0) statsSm[lx] = rsum;

  // normalize head output and store (bf16, token-major for the O-proj GEMM)
#pragma unroll
  for (int r = 0; r < 8; ++r) {
    const float inv = 1.0f / statsSm[half * 8 + r];
    acc[0][r] *= inv; acc[1][r] *= inv; acc[2][r] *= inv; acc[3][r] *= inv;
  }
#pragma unroll
  for (int nt = 0; nt < 4; ++nt) {
    const int ncol = h * HDIM + nt * 16 + lx;
#pragma unroll
    for (int r = 0; r < 8; ++r) {
      const int tok = (b << 11) + q0 + r + half * 8;
      OutH[(size_t)tok * DD + ncol] = (__bf16)acc[nt][r];
    }
  }

  // ---- pass 3: single fp32 write of the full 16x2048 probability strip ---
  float* dst = attn_out + ((size_t)bh << 22) + ((size_t)q0 << 11);
  for (int it = 0; it < 256; ++it) {
    const int eid = it * 32 + lane;   // float4 units: 16 rows * 512
    const int row = eid >> 9;
    const int col = (eid & 511) << 2;
    float4 o;
    if (col < nkeys) {
      const float inv = 1.0f / statsSm[row];
      const float4 p = *(float4*)&S[row * SROW + col];
      o = make_float4(p.x * inv, p.y * inv, p.z * inv, p.w * inv);
    } else {
      o = make_float4(0.f, 0.f, 0.f, 0.f);   // causal tail: explicit zeros
    }
    *(float4*)&dst[((size_t)row << 11) + col] = o;
  }
}

// ---------------------------------------------------------------------------
// Kernel 3: output projection  out = OutH @ Wo^T + bo   (fp32 out)
// ---------------------------------------------------------------------------
__global__ void hca_oproj_kernel(const __bf16* __restrict__ OutH,
                                 const float* __restrict__ Wo,
                                 const float* __restrict__ bo,
                                 float* __restrict__ out) {
  const int lane = threadIdx.x & 31;
  const int wave = threadIdx.x >> 5;
  const int m0 = blockIdx.y * 16;
  const int n0 = (blockIdx.x * 8 + wave) * 16;
  const int half = lane >> 4, lx = lane & 15;
  const int kbase = half * 8;

  const __bf16* arow = OutH + (size_t)(m0 + lx) * DD;
  const float*  brow = Wo   + (size_t)(n0 + lx) * DD;

  v8f acc = {};
  for (int kt = 0; kt < DD; kt += 32) {
    v16bf A = ld16_bf(arow + kt + kbase, arow + kt + 16 + kbase);
    v16bf B = ld16_f32(brow + kt + kbase, brow + kt + 16 + kbase);
    acc = wmma_bf16(A, B, acc);
  }
  const int ncol = n0 + lx;
  const float add = bo[ncol];
#pragma unroll
  for (int r = 0; r < 8; ++r) {
    const int tok = m0 + r + half * 8;
    out[(size_t)tok * DD + ncol] = acc[r] + add;
  }
}

// ---------------------------------------------------------------------------
extern "C" void kernel_launch(void* const* d_in, const int* in_sizes, int n_in,
                              void* d_out, int out_size, void* d_ws, size_t ws_size,
                              hipStream_t stream) {
  (void)in_sizes; (void)n_in; (void)out_size; (void)ws_size;

  const float* x  = (const float*)d_in[0];
  const float* fb = (const float*)d_in[1];
  const float* Wq = (const float*)d_in[2];
  const float* bq = (const float*)d_in[3];
  const float* Wk = (const float*)d_in[4];
  const float* bk = (const float*)d_in[5];
  const float* Wv = (const float*)d_in[6];
  const float* bv = (const float*)d_in[7];
  const float* Wo = (const float*)d_in[8];
  const float* bo = (const float*)d_in[9];

  float* out  = (float*)d_out;                        // (B,L,D)
  float* attn = out + (size_t)4194304;                // (B,H,L,L)
  float* vout = attn + (size_t)134217728;             // (B,H,L,hd)

  __bf16* Qws  = (__bf16*)d_ws;                       // (B*L, D) bf16
  __bf16* Kws  = Qws + (size_t)4194304;               // (B*L, D) bf16
  __bf16* Vtws = Kws + (size_t)4194304;               // (B,H,hd,L) bf16
  __bf16* OutH = Vtws + (size_t)4194304;              // (B*L, D) bf16

  // 1) QKV projections (+feedback on Q); V also emitted fp32 to d_out
  hca_qkv_kernel<<<dim3(8, 256, 3), 256, 0, stream>>>(
      x, fb, Wq, bq, Wk, bk, Wv, bv, Qws, Kws, Vtws, vout);

  // 2) attention: 4096 query tiles, 2 waves per block, 257 KB LDS / block
  const size_t smem_bytes = (2 * 16 * SROW + 64) * sizeof(float);
  hca_attn_kernel<<<dim3(2048), 64, smem_bytes, stream>>>(
      Qws, Kws, Vtws, OutH, attn);

  // 3) output projection
  hca_oproj_kernel<<<dim3(8, 256, 1), 256, 0, stream>>>(OutH, Wo, bo, out);
}
